// MultiheadAttention_30932354465831
// MI455X (gfx1250) — compile-verified, matrix operand order checked by reading, not run on hardware
//
#include <hip/hip_runtime.h>
#include <hip/hip_bf16.h>

// ---------------- types ----------------
typedef __bf16 bf16;
typedef __attribute__((ext_vector_type(16))) __bf16 v16bf;
typedef __attribute__((ext_vector_type(8)))  __bf16 v8bf;
typedef __attribute__((ext_vector_type(4)))  __bf16 v4bf;
typedef __attribute__((ext_vector_type(8)))  float  v8f;

// ---------------- problem constants ----------------
#define S_LEN   2048
#define BATCH   2
#define DMODEL  768
#define NHEAD   12
#define HDIM    64
#define RLORA   64
#define MROWS   (S_LEN*BATCH)   // 4096
#define N3      (3*DMODEL)      // 2304

// ---------------- GEMM tiling ----------------
#define KT   32     // K per WMMA step
#define LDK  40     // padded LDS stride (elements); 80B, keeps 16B alignment
#define BM   128    // block rows  (4 wave-rows of 32)
#define BN   64     // block cols  (2 wave-cols of 32)
#define NT   256    // 8 wave32

// Combine two 8-element bf16 LDS chunks into the 16-element WMMA operand.
__device__ __forceinline__ v16bf bf16x16(const bf16* p) {
  v8bf a = *(const v8bf*)p;
  v8bf b = *(const v8bf*)(p + 16);
  return __builtin_shufflevector(a, b, 0,1,2,3,4,5,6,7,8,9,10,11,12,13,14,15);
}

__device__ __forceinline__ v8f wmma_bf16(v16bf a, v16bf b, v8f c) {
  return __builtin_amdgcn_wmma_f32_16x16x32_bf16(false, a, false, b, (short)0, c, false, false);
}

__device__ __forceinline__ v4bf cvt4(float4 f) {
  v4bf h = { (bf16)f.x, (bf16)f.y, (bf16)f.z, (bf16)f.w };
  return h;
}

// GEMM macro-tile, double-buffered LDS pipeline.
// acc[4] += A[rowBase:+128, :K] * W[colBase:+64, :K]^T ; wave = 32x32 output.
// As sized 2*BM*LDK, Bs sized 2*BN*LDK.
__device__ __forceinline__ void gemm_bf16_tile(
    const float* __restrict__ A, int ldA,
    const float* __restrict__ W, int ldW,
    int K, int rowBase, int colBase,
    bf16* As, bf16* Bs, v8f acc[4])
{
  const int tid  = threadIdx.x;
  const int lane = tid & 31, wave = tid >> 5;
  const int wm   = wave >> 1, wn = wave & 1;
  const int half = lane >> 4, l15 = lane & 15;

  // per-thread staging plan (hoisted): A 128x32 = 1024 float4 (4/thr), B 64x32 = 512 (2/thr)
  const int c4 = (tid & 7) << 2;
  const int rA = tid >> 3;
  size_t aoff[4]; int asoff[4];
  #pragma unroll
  for (int i = 0; i < 4; ++i) {
    int r = rA + i*32;
    aoff[i]  = (size_t)(rowBase + r)*ldA + c4;
    asoff[i] = r*LDK + c4;
  }
  size_t boff[2]; int bsoff[2];
  #pragma unroll
  for (int i = 0; i < 2; ++i) {
    int r = rA + i*32;
    boff[i]  = (size_t)(colBase + r)*ldW + c4;
    bsoff[i] = r*LDK + c4;
  }

  const int kb  = half * 8;
  const int ar0 = wm*32 + l15;
  const int br0 = wn*32 + l15;

  __syncthreads();                         // protect LDS against prior use
  // preload tile 0 into buffer 0
  #pragma unroll
  for (int i = 0; i < 4; ++i) *(v4bf*)(As + asoff[i]) = cvt4(*(const float4*)(A + aoff[i]));
  #pragma unroll
  for (int i = 0; i < 2; ++i) *(v4bf*)(Bs + bsoff[i]) = cvt4(*(const float4*)(W + boff[i]));

  int cur = 0;
  for (int k0 = 0; k0 < K; k0 += KT) {
    __syncthreads();                       // buf[cur] ready; everyone done with buf[cur^1]
    const bool hasNext = (k0 + KT) < K;
    v4bf ra[4], rb[2];
    if (hasNext) {                         // issue next-tile loads before the WMMAs
      #pragma unroll
      for (int i = 0; i < 4; ++i) ra[i] = cvt4(*(const float4*)(A + aoff[i] + k0 + KT));
      #pragma unroll
      for (int i = 0; i < 2; ++i) rb[i] = cvt4(*(const float4*)(W + boff[i] + k0 + KT));
      if (k0 + 2*KT < K) {                 // deep prefetch -> global_prefetch_b8
        __builtin_prefetch(A + aoff[0] + k0 + 2*KT, 0, 1);
        __builtin_prefetch(W + boff[0] + k0 + 2*KT, 0, 1);
      }
    }
    const bf16* Ac = As + cur*(BM*LDK);
    const bf16* Bc = Bs + cur*(BN*LDK);
    v16bf a0 = bf16x16(&Ac[ ar0      *LDK + kb]);
    v16bf a1 = bf16x16(&Ac[(ar0 + 16)*LDK + kb]);
    v16bf b0 = bf16x16(&Bc[ br0      *LDK + kb]);
    v16bf b1 = bf16x16(&Bc[(br0 + 16)*LDK + kb]);
    acc[0] = wmma_bf16(a0, b0, acc[0]);
    acc[1] = wmma_bf16(a0, b1, acc[1]);
    acc[2] = wmma_bf16(a1, b0, acc[2]);
    acc[3] = wmma_bf16(a1, b1, acc[3]);
    if (hasNext) {
      bf16* An = As + (cur^1)*(BM*LDK);
      bf16* Bn = Bs + (cur^1)*(BN*LDK);
      #pragma unroll
      for (int i = 0; i < 4; ++i) *(v4bf*)(An + asoff[i]) = ra[i];
      #pragma unroll
      for (int i = 0; i < 2; ++i) *(v4bf*)(Bn + bsoff[i]) = rb[i];
    }
    cur ^= 1;
  }
}

// ---------------- kernel 1/4: LoRA down:  H = relu(X * DW^T + DB) ----------------
__global__ void __launch_bounds__(NT)
k_lora_down(const float* __restrict__ X, const float* __restrict__ DW,
            const float* __restrict__ DB, float* __restrict__ Hout)
{
  __shared__ __align__(16) bf16 As[2*BM*LDK];
  __shared__ __align__(16) bf16 Bs[2*BN*LDK];
  const int rowBase = blockIdx.x * BM;   // colBase = 0 (R == BN)
  v8f acc[4] = {{}, {}, {}, {}};
  gemm_bf16_tile(X, DMODEL, DW, DMODEL, DMODEL, rowBase, 0, As, Bs, acc);

  const int lane = threadIdx.x & 31, wave = threadIdx.x >> 5;
  const int wm = wave >> 1, wn = wave & 1, half = lane >> 4, l15 = lane & 15;
  #pragma unroll
  for (int ni = 0; ni < 2; ++ni) {
    const int col = wn*32 + ni*16 + l15;
    const float db = DB[col];
    #pragma unroll
    for (int mi = 0; mi < 2; ++mi) {
      #pragma unroll
      for (int i = 0; i < 8; ++i) {
        int row = rowBase + wm*32 + mi*16 + i + half*8;
        float y = acc[mi*2 + ni][i] + db;
        Hout[(size_t)row*RLORA + col] = y > 0.f ? y : 0.f;
      }
    }
  }
}

// ---------------- kernel 2: fused QKV proj + SSF + LoRA1-up ----------------
// q,k -> [B*H][S][64] bf16 ; v -> transposed [B*H][64][S] bf16 (for barrier-free attn staging)
__global__ void __launch_bounds__(NT)
k_qkv(const float* __restrict__ X,  const float* __restrict__ W,  const float* __restrict__ Wb,
      const float* __restrict__ sc1, const float* __restrict__ sh1,
      const float* __restrict__ H1, const float* __restrict__ UW, const float* __restrict__ UB,
      const float* __restrict__ lsc,
      bf16* __restrict__ Q, bf16* __restrict__ Kh, bf16* __restrict__ Vt)
{
  __shared__ __align__(16) bf16 As[2*BM*LDK];
  __shared__ __align__(16) bf16 Bs[2*BN*LDK];
  const int rowBase = blockIdx.x * BM, colBase = blockIdx.y * BN;
  v8f acc[4]  = {{}, {}, {}, {}};
  v8f lacc[4] = {{}, {}, {}, {}};
  gemm_bf16_tile(X,  DMODEL, W,  DMODEL, DMODEL, rowBase, colBase, As, Bs, acc);
  gemm_bf16_tile(H1, RLORA,  UW, RLORA,  RLORA,  rowBase, colBase, As, Bs, lacc);

  const int lane = threadIdx.x & 31, wave = threadIdx.x >> 5;
  const int wm = wave >> 1, wn = wave & 1, half = lane >> 4, l15 = lane & 15;
  const float ls = lsc[0];
  #pragma unroll
  for (int ni = 0; ni < 2; ++ni) {
    const int e = colBase + wn*32 + ni*16 + l15;   // 0..2303
    const float bb = Wb[e], s1 = sc1[e], h1 = sh1[e], ub = UB[e];
    const int sec  = e / DMODEL;                   // 0=q 1=k 2=v
    const int dcol = e % DMODEL;
    const int head = dcol >> 6, j = dcol & 63;
    const float mul = (sec == 0) ? 0.125f : 1.0f;  // hd^-0.5
    #pragma unroll
    for (int mi = 0; mi < 2; ++mi) {
      #pragma unroll
      for (int i = 0; i < 8; ++i) {
        int m = rowBase + wm*32 + mi*16 + i + half*8;  // m = s*B + b
        int s = m >> 1, bidx = m & 1;
        int n = bidx*NHEAD + head;
        float y = (acc[mi*2+ni][i] + bb)*s1 + h1 + (lacc[mi*2+ni][i] + ub)*ls;
        bf16 hv = (bf16)(y * mul);
        if (sec == 2)      Vt[((size_t)n*HDIM + j)*S_LEN + s] = hv;       // transposed
        else if (sec == 1) Kh[((size_t)n*S_LEN + s)*HDIM + j] = hv;
        else               Q [((size_t)n*S_LEN + s)*HDIM + j] = hv;
      }
    }
  }
}

// ---------------- kernel 3: flash attention, double-buffered KV, 64-key tiles ----------------
// grid: (S/128, B*H); block 256. Each wave owns 16 query rows.
#define TKV 64
#define LQ  72   // Qs/Ks/Vs/Ps padded stride (144B, 16B-aligned chunks)
__global__ void __launch_bounds__(NT)
k_attn(const bf16* __restrict__ Q, const bf16* __restrict__ Kh,
       const bf16* __restrict__ Vt, float* __restrict__ Aout)
{
  const int n  = blockIdx.y;
  const int q0 = blockIdx.x * 128;
  const int tid = threadIdx.x, lane = tid & 31, wave = tid >> 5;
  const int half = lane >> 4, l15 = lane & 15;

  __shared__ __align__(16) bf16 Qs[128*LQ];        // 18 KB
  __shared__ __align__(16) bf16 Ks[2*TKV*LQ];      // [key][dim], double buffered
  __shared__ __align__(16) bf16 Vs[2*HDIM*LQ];     // [dim][key], double buffered
  __shared__ __align__(16) bf16 Ps[8*16*LQ];       // per-wave P tile (16 x 64 keys)

  const bf16* qn = Q  + (size_t)n*S_LEN*HDIM;
  const bf16* kn = Kh + (size_t)n*S_LEN*HDIM;
  const bf16* vn = Vt + (size_t)n*HDIM*S_LEN;

  // stage 128x64 Q strip once: 16B chunks, 4/thread
  const int c8 = (tid & 7) << 3;
  const int rT = tid >> 3;
  #pragma unroll
  for (int i = 0; i < 4; ++i) {
    int r = rT + i*32;
    *(v8bf*)&Qs[r*LQ + c8] = *(const v8bf*)&qn[(size_t)(q0 + r)*HDIM + c8];
  }

  // KV staging plan: K tile 64x64 (2 chunks/thr), V tile 64x64 transposed source (2 chunks/thr)
  size_t kgo[2], vgo[2]; int kso[2], vso[2];
  #pragma unroll
  for (int i = 0; i < 2; ++i) {
    int r = rT + i*32;
    kgo[i] = (size_t)r*HDIM + c8;        // + t0*HDIM per tile (key rows advance)
    kso[i] = r*LQ + c8;
    vgo[i] = (size_t)r*S_LEN + c8;       // r = dim, c8 = key offset; + t0 per tile
    vso[i] = r*LQ + c8;
  }
  // preload tile 0 into buffer 0
  #pragma unroll
  for (int i = 0; i < 2; ++i) {
    *(v8bf*)&Ks[kso[i]] = *(const v8bf*)&kn[kgo[i]];
    *(v8bf*)&Vs[vso[i]] = *(const v8bf*)&vn[vgo[i]];
  }

  v8f o0 = {}, o1 = {}, o2 = {}, o3 = {};
  float mrow[8], lrow[8];
  #pragma unroll
  for (int i = 0; i < 8; ++i) { mrow[i] = -1e30f; lrow[i] = 0.f; }

  bf16* pw        = &Ps[wave*16*LQ];
  const bf16* prd = &Ps[wave*16*LQ + l15*LQ];
  const int arow  = wave*16 + l15;
  const int kbh   = half*8;

  int cur = 0;
  for (int t0 = 0; t0 < S_LEN; t0 += TKV) {
    __syncthreads();                         // buf[cur] ready; buf[cur^1] free
    const bool hasNext = (t0 + TKV) < S_LEN;
    v8bf rk[2], rv[2];
    if (hasNext) {
      #pragma unroll
      for (int i = 0; i < 2; ++i) {
        rk[i] = *(const v8bf*)&kn[kgo[i] + (size_t)(t0 + TKV)*HDIM];
        rv[i] = *(const v8bf*)&vn[vgo[i] + (size_t)(t0 + TKV)];
      }
    }
    const bf16* Kc = Ks + cur*(TKV*LQ);
    const bf16* Vc = Vs + cur*(HDIM*LQ);

    // scores: [16q x 64keys] = Q(16x64) * K^T(64x64)
    v8f sj[4] = {{}, {}, {}, {}};
    #pragma unroll
    for (int kk = 0; kk < HDIM; kk += KT) {
      v16bf aq = bf16x16(&Qs[arow*LQ + kk + kbh]);
      #pragma unroll
      for (int j = 0; j < 4; ++j)
        sj[j] = wmma_bf16(aq, bf16x16(&Kc[(j*16 + l15)*LQ + kk + kbh]), sj[j]);
    }

    // online softmax; row = i + 8*half; reductions stay in 16-lane halves
    #pragma unroll
    for (int i = 0; i < 8; ++i) {
      float mx = fmaxf(fmaxf(sj[0][i], sj[1][i]), fmaxf(sj[2][i], sj[3][i]));
      #pragma unroll
      for (int off = 8; off >= 1; off >>= 1) mx = fmaxf(mx, __shfl_xor(mx, off, 32));
      float mnew  = fmaxf(mrow[i], mx);
      float alpha = __expf(mrow[i] - mnew);
      mrow[i] = mnew;
      float p0 = __expf(sj[0][i] - mnew);
      float p1 = __expf(sj[1][i] - mnew);
      float p2 = __expf(sj[2][i] - mnew);
      float p3 = __expf(sj[3][i] - mnew);
      float sum = (p0 + p1) + (p2 + p3);
      #pragma unroll
      for (int off = 8; off >= 1; off >>= 1) sum += __shfl_xor(sum, off, 32);
      lrow[i] = lrow[i]*alpha + sum;
      o0[i] *= alpha; o1[i] *= alpha; o2[i] *= alpha; o3[i] *= alpha;
      int prow = i + half*8;
      pw[prow*LQ +  0 + l15] = (bf16)p0;
      pw[prow*LQ + 16 + l15] = (bf16)p1;
      pw[prow*LQ + 32 + l15] = (bf16)p2;
      pw[prow*LQ + 48 + l15] = (bf16)p3;
    }

    // P(16x64) * V(64x64): two K-steps of 32 keys each
    v16bf ap0 = bf16x16(&prd[kbh]);
    v16bf ap1 = bf16x16(&prd[32 + kbh]);
    o0 = wmma_bf16(ap0, bf16x16(&Vc[( 0 + l15)*LQ + kbh]), o0);
    o1 = wmma_bf16(ap0, bf16x16(&Vc[(16 + l15)*LQ + kbh]), o1);
    o2 = wmma_bf16(ap0, bf16x16(&Vc[(32 + l15)*LQ + kbh]), o2);
    o3 = wmma_bf16(ap0, bf16x16(&Vc[(48 + l15)*LQ + kbh]), o3);
    o0 = wmma_bf16(ap1, bf16x16(&Vc[( 0 + l15)*LQ + 32 + kbh]), o0);
    o1 = wmma_bf16(ap1, bf16x16(&Vc[(16 + l15)*LQ + 32 + kbh]), o1);
    o2 = wmma_bf16(ap1, bf16x16(&Vc[(32 + l15)*LQ + 32 + kbh]), o2);
    o3 = wmma_bf16(ap1, bf16x16(&Vc[(48 + l15)*LQ + 32 + kbh]), o3);

    if (hasNext) {
      bf16* Kn = Ks + (cur^1)*(TKV*LQ);
      bf16* Vn = Vs + (cur^1)*(HDIM*LQ);
      #pragma unroll
      for (int i = 0; i < 2; ++i) {
        *(v8bf*)&Kn[kso[i]] = rk[i];
        *(v8bf*)&Vn[vso[i]] = rv[i];
      }
    }
    cur ^= 1;
  }

  // epilogue: normalize, scatter back to [S, B, D]
  const int b = n / NHEAD, h = n % NHEAD;
  #pragma unroll
  for (int i = 0; i < 8; ++i) {
    float inv = 1.f / lrow[i];
    int srow = q0 + wave*16 + i + half*8;
    size_t base = (size_t)(srow*BATCH + b)*DMODEL + h*HDIM + l15;
    Aout[base +  0] = o0[i]*inv;
    Aout[base + 16] = o1[i]*inv;
    Aout[base + 32] = o2[i]*inv;
    Aout[base + 48] = o3[i]*inv;
  }
}

// ---------------- kernel 5: fused out-proj + SSF2 + LoRA2-up ----------------
__global__ void __launch_bounds__(NT)
k_outproj(const float* __restrict__ X,  const float* __restrict__ W,  const float* __restrict__ Wb,
          const float* __restrict__ sc2, const float* __restrict__ sh2,
          const float* __restrict__ H2, const float* __restrict__ UW, const float* __restrict__ UB,
          const float* __restrict__ lsc, float* __restrict__ Out)
{
  __shared__ __align__(16) bf16 As[2*BM*LDK];
  __shared__ __align__(16) bf16 Bs[2*BN*LDK];
  const int rowBase = blockIdx.x * BM, colBase = blockIdx.y * BN;
  v8f acc[4]  = {{}, {}, {}, {}};
  v8f lacc[4] = {{}, {}, {}, {}};
  gemm_bf16_tile(X,  DMODEL, W,  DMODEL, DMODEL, rowBase, colBase, As, Bs, acc);
  gemm_bf16_tile(H2, RLORA,  UW, RLORA,  RLORA,  rowBase, colBase, As, Bs, lacc);

  const int lane = threadIdx.x & 31, wave = threadIdx.x >> 5;
  const int wm = wave >> 1, wn = wave & 1, half = lane >> 4, l15 = lane & 15;
  const float ls = lsc[0];
  #pragma unroll
  for (int ni = 0; ni < 2; ++ni) {
    const int e = colBase + wn*32 + ni*16 + l15;
    const float bb = Wb[e], s2 = sc2[e], h2 = sh2[e], ub = UB[e];
    #pragma unroll
    for (int mi = 0; mi < 2; ++mi) {
      #pragma unroll
      for (int i = 0; i < 8; ++i) {
        int m = rowBase + wm*32 + mi*16 + i + half*8;
        float y = (acc[mi*2+ni][i] + bb)*s2 + h2 + (lacc[mi*2+ni][i] + ub)*ls;
        Out[(size_t)m*DMODEL + e] = y;
      }
    }
  }
}

// ---------------- host launcher ----------------
extern "C" void kernel_launch(void* const* d_in, const int* in_sizes, int n_in,
                              void* d_out, int out_size, void* d_ws, size_t ws_size,
                              hipStream_t stream) {
  (void)in_sizes; (void)n_in; (void)out_size; (void)ws_size;
  const float* query = (const float*)d_in[0];
  // d_in[1]=key, d_in[2]=value: unused by the reference math (qkv all from query)
  const float* in_w  = (const float*)d_in[3];
  const float* in_b  = (const float*)d_in[4];
  const float* sc1   = (const float*)d_in[5];
  const float* sh1   = (const float*)d_in[6];
  const float* l1dw  = (const float*)d_in[7];
  const float* l1db  = (const float*)d_in[8];
  const float* l1uw  = (const float*)d_in[9];
  const float* l1ub  = (const float*)d_in[10];
  const float* l1sc  = (const float*)d_in[11];
  const float* ow    = (const float*)d_in[12];
  const float* ob    = (const float*)d_in[13];
  const float* sc2   = (const float*)d_in[14];
  const float* sh2   = (const float*)d_in[15];
  const float* l2dw  = (const float*)d_in[16];
  const float* l2db  = (const float*)d_in[17];
  const float* l2uw  = (const float*)d_in[18];
  const float* l2ub  = (const float*)d_in[19];
  const float* l2sc  = (const float*)d_in[20];

  // workspace layout (32 MiB total)
  char* ws = (char*)d_ws;
  float* h1   = (float*)(ws);                                   // 4096*64 f32   = 1 MiB
  bf16*  qb   = (bf16*) (ws + (1u<<20));                        // 24*2048*64    = 6 MiB
  bf16*  kb   = (bf16*) (ws + (1u<<20) + 6291456u);
  bf16*  vtb  = (bf16*) (ws + (1u<<20) + 2u*6291456u);          // V transposed
  float* attn = (float*)(ws + (1u<<20) + 3u*6291456u);          // 4096*768 f32  = 12 MiB
  float* h2   = (float*)(ws + (1u<<20) + 3u*6291456u + 12582912u);
  float* out  = (float*)d_out;

  dim3 blk(NT);
  // 1) LoRA1 down
  k_lora_down<<<dim3(MROWS/BM, 1), blk, 0, stream>>>(query, l1dw, l1db, h1);
  // 2) fused QKV + SSF + LoRA1 up; q,k -> [B*H,S,64]; v -> [B*H,64,S]
  k_qkv<<<dim3(MROWS/BM, N3/BN), blk, 0, stream>>>(query, in_w, in_b, sc1, sh1,
                                                   h1, l1uw, l1ub, l1sc, qb, kb, vtb);
  // 3) flash attention over 24 head-batches
  k_attn<<<dim3(S_LEN/128, BATCH*NHEAD), blk, 0, stream>>>(qb, kb, vtb, attn);
  // 4) LoRA2 down on attention output
  k_lora_down<<<dim3(MROWS/BM, 1), blk, 0, stream>>>(attn, l2dw, l2db, h2);
  // 5) fused out-proj + SSF2 + LoRA2 up
  k_outproj<<<dim3(MROWS/BM, DMODEL/BN), blk, 0, stream>>>(attn, ow, ob, sc2, sh2,
                                                           h2, l2uw, l2ub, l2sc, out);
}
